// TextureSynthesizer_21337397527043
// MI455X (gfx1250) — matrix operand order repaired and checked by reading, not compiled, not run on hardware
//
#include <hip/hip_runtime.h>
#include <hip/hip_bf16.h>
#include <stdint.h>

// Problem constants (match reference)
#define BATCH 32
#define CHAN  256
#define NSP   4096              // H*W
#define KTOP  409               // int(0.1 * 4096)
#define NROWS (BATCH * CHAN)    // 8192

// Gram-stage tiling
#define AROWS 128               // G rows per block
#define BROWS 64                // G cols per block
#define KC    64                // K chunk staged in LDS
#define LDS_S 72                // padded LDS row stride (bf16 elems) -> conflict-free b128 reads

typedef __attribute__((ext_vector_type(16))) __bf16        v16bf;
typedef __attribute__((ext_vector_type(8)))  float         v8f;
typedef __attribute__((ext_vector_type(4)))  unsigned int  v4u;

// ---------------------------------------------------------------------------
// Stage 1: per-row exact top-k mask via 4-pass radix select on bits of |x|,
// write sparsified activations as bf16 into workspace.
// One 256-thread block per (b,c) row of 4096 floats.
// ---------------------------------------------------------------------------
__global__ __launch_bounds__(256)
void topk_mask_kernel(const float* __restrict__ x,
                      __hip_bfloat16* __restrict__ F) {
  __shared__ float         xs[NSP];      // 16 KB row cache
  __shared__ unsigned int  hist[256];    // radix histogram
  __shared__ unsigned int  sel[2];       // [0]=prefix (kth key), [1]=remaining k
  __shared__ unsigned int  eqctr;        // tie apportioning counter

  const int row = blockIdx.x;
  const int tid = threadIdx.x;
  const float* xr = x + (size_t)row * NSP;

  for (int i = tid; i < NSP; i += 256) xs[i] = xr[i];
  if (tid == 0) { sel[0] = 0u; sel[1] = KTOP; eqctr = 0u; }
  __syncthreads();

  // 4 radix passes, MSB byte -> LSB byte. Key = bits of |x| (monotone, bit31=0).
  for (int pass = 0; pass < 4; ++pass) {
    const int shift = 24 - pass * 8;
    hist[tid] = 0u;
    __syncthreads();
    const unsigned int pfx = sel[0];
    const unsigned int msk = (pass == 0) ? 0u : ~((1u << (shift + 8)) - 1u);
    for (int i = tid; i < NSP; i += 256) {
      unsigned int key = __float_as_uint(fabsf(xs[i]));
      if ((key & msk) == pfx) {
        atomicAdd(&hist[(key >> shift) & 0xFFu], 1u);
      }
    }
    __syncthreads();
    if (tid == 0) {
      unsigned int kk = sel[1];
      unsigned int cum = 0u;
      int bsel = 0;
      for (int bin = 255; bin >= 0; --bin) {
        unsigned int c = hist[bin];
        if (cum + c >= kk) { bsel = bin; break; }
        cum += c;
      }
      sel[0] |= ((unsigned int)bsel) << shift;
      sel[1] = kk - cum;   // how many equal-key elements still needed
    }
    __syncthreads();
  }

  const unsigned int kth = sel[0];   // exact k-th largest |x| bit pattern
  const unsigned int kk  = sel[1];   // number of ties to keep
  __hip_bfloat16* Fr = F + (size_t)row * NSP;
  for (int i = tid; i < NSP; i += 256) {
    float v = xs[i];
    unsigned int key = __float_as_uint(fabsf(v));
    bool keep;
    if (key > kth)       keep = true;
    else if (key == kth) keep = (atomicAdd(&eqctr, 1u) < kk);
    else                 keep = false;
    Fr[i] = __float2bfloat16(keep ? v : 0.0f);
  }
}

// ---------------------------------------------------------------------------
// Stage 2 helpers
// ---------------------------------------------------------------------------
// One 16-byte async global->LDS copy (CDNA5 ASYNCcnt path).
__device__ __forceinline__ void async_ld16(const __hip_bfloat16* src,
                                           const __hip_bfloat16* dstLds) {
  // Generic pointers into LDS carry the LDS byte offset in their low 32 bits.
  asm volatile("global_load_async_to_lds_b128 %0, %1, off"
               :: "v"((uint32_t)(uintptr_t)dstLds), "v"((const void*)src)
               : "memory");
}

// A fragment (16x32, MxK): lane = row M (lane%16); VGPR0-3 <- K=[h*8..h*8+7],
// VGPR4-7 <- K=[16+h*8..16+h*8+7].
__device__ __forceinline__ v16bf lds_frag_A(const __hip_bfloat16* base,
                                            int row, int kk, int half) {
  union { v4u u[2]; v16bf v; } a;
  a.u[0] = *(const v4u*)(base + row * LDS_S + kk + half * 8);
  a.u[1] = *(const v4u*)(base + row * LDS_S + kk + 16 + half * 8);
  return a.v;
}
// B fragment (32x16, KxN): lane = col N (lane%16); VGPR0-3 <- K=[h*8..h*8+7], VGPR4-7 <- K=[16+h*8..16+h*8+7].
__device__ __forceinline__ v16bf lds_frag_B(const __hip_bfloat16* base,
                                            int row, int kk, int half) {
  union { v4u u[2]; v16bf v; } b;
  b.u[0] = *(const v4u*)(base + row * LDS_S + kk + half * 8);
  b.u[1] = *(const v4u*)(base + row * LDS_S + kk + 16 + half * 8);
  return b.v;
}

// ---------------------------------------------------------------------------
// Stage 2: batched Gram G = F F^T via v_wmma_f32_16x16x32_bf16, fused with
// loss = 1e9 * mean((target - G/(B*C*H*W))^2).
// Block = 8 waves computes a 128x64 tile of G for one batch; K staged through
// double-buffered LDS with global_load_async_to_lds_b128 (ASYNCcnt).
// ---------------------------------------------------------------------------
__global__ __launch_bounds__(256)
void gram_loss_kernel(const __hip_bfloat16* __restrict__ F,
                      const float* __restrict__ target,
                      float* __restrict__ out) {
  __shared__ alignas(16) __hip_bfloat16 aSh[2][AROWS * LDS_S];  // 2 x 18 KB
  __shared__ alignas(16) __hip_bfloat16 bSh[2][BROWS * LDS_S];  // 2 x  9 KB

  const int tid  = threadIdx.x;
  const int wave = tid >> 5;
  const int lane = tid & 31;
  const int half = lane >> 4;
  const int l16  = lane & 15;

  const int b   = blockIdx.x >> 3;      // batch
  const int blk = blockIdx.x & 7;       // 2 x 4 block grid over 256x256 G
  const int i0  = (blk >> 2) * AROWS;   // 0 or 128
  const int j0  = (blk & 3) * BROWS;    // 0,64,128,192

  const __hip_bfloat16* Fb = F + (size_t)b * CHAN * NSP;

  const int wi = wave >> 1;             // 0..3 : 32-row strip within block
  const int wj = wave & 1;              // 0..1 : 32-col strip within block

  // Cooperative async stage of one K-chunk into LDS buffer `buf`.
  auto stage = [&](int k0, int buf) {
#pragma unroll
    for (int i = 0; i < 4; ++i) {                // A: 128 rows x 8 vec = 1024
      int v = tid + i * 256;
      int row = v >> 3, c = v & 7;
      async_ld16(Fb + (size_t)(i0 + row) * NSP + k0 + c * 8,
                 &aSh[buf][row * LDS_S + c * 8]);
    }
#pragma unroll
    for (int i = 0; i < 2; ++i) {                // B: 64 rows x 8 vec = 512
      int v = tid + i * 256;
      int row = v >> 3, c = v & 7;
      async_ld16(Fb + (size_t)(j0 + row) * NSP + k0 + c * 8,
                 &bSh[buf][row * LDS_S + c * 8]);
    }
  };

  v8f acc[2][2] = {};

  stage(0, 0);                                   // prologue fill
  const int NCHUNK = NSP / KC;                   // 64
  for (int c = 0; c < NCHUNK; ++c) {
    const int cur = c & 1;
    if (c + 1 < NCHUNK) {
      stage((c + 1) * KC, cur ^ 1);              // overlap next fill with compute
      // 6 loads/thread outstanding for the next buffer; drain older ones only.
      asm volatile("s_wait_asynccnt 0x6" ::: "memory");
    } else {
      asm volatile("s_wait_asynccnt 0x0" ::: "memory");
    }
    __syncthreads();

    const __hip_bfloat16* aB = &aSh[cur][0];
    const __hip_bfloat16* bB = &bSh[cur][0];
#pragma unroll
    for (int kk = 0; kk < KC; kk += 32) {
      v16bf a0 = lds_frag_A(aB, wi * 32 + l16,      kk, half);
      v16bf a1 = lds_frag_A(aB, wi * 32 + 16 + l16, kk, half);
      v16bf b0 = lds_frag_B(bB, wj * 32 + l16,      kk, half);
      v16bf b1 = lds_frag_B(bB, wj * 32 + 16 + l16, kk, half);
      acc[0][0] = __builtin_amdgcn_wmma_f32_16x16x32_bf16(false, a0, false, b0,
                                                          (short)0, acc[0][0], false, false);
      acc[0][1] = __builtin_amdgcn_wmma_f32_16x16x32_bf16(false, a0, false, b1,
                                                          (short)0, acc[0][1], false, false);
      acc[1][0] = __builtin_amdgcn_wmma_f32_16x16x32_bf16(false, a1, false, b0,
                                                          (short)0, acc[1][0], false, false);
      acc[1][1] = __builtin_amdgcn_wmma_f32_16x16x32_bf16(false, a1, false, b1,
                                                          (short)0, acc[1][1], false, false);
    }
    __syncthreads();                             // all waves done before overwrite
  }

  // Epilogue: C/D layout is VGPR r -> row r + 8*(lane/16), col = lane%16.
  const float inv_norm   = 1.0f / (float)(BATCH * CHAN * NSP);     // 1/33554432
  const float loss_scale = 1.0e9f / (float)(BATCH * CHAN * CHAN);  // 1e9 / mean count
  const float* tgb = target + (size_t)b * CHAN * CHAN;
  float s = 0.0f;
#pragma unroll
  for (int ti = 0; ti < 2; ++ti) {
#pragma unroll
    for (int tj = 0; tj < 2; ++tj) {
#pragma unroll
      for (int r = 0; r < 8; ++r) {
        int row = i0 + wi * 32 + ti * 16 + r + half * 8;
        int col = j0 + wj * 32 + tj * 16 + l16;
        float g = acc[ti][tj][r] * inv_norm;
        float d = tgb[row * CHAN + col] - g;
        s += d * d;
      }
    }
  }
  s *= loss_scale;
#pragma unroll
  for (int off = 16; off > 0; off >>= 1) s += __shfl_xor(s, off, 32);
  if (lane == 0) atomicAdd(out, s);
}

// ---------------------------------------------------------------------------
extern "C" void kernel_launch(void* const* d_in, const int* in_sizes, int n_in,
                              void* d_out, int out_size, void* d_ws, size_t ws_size,
                              hipStream_t stream) {
  const float* x      = (const float*)d_in[0];       // (32,256,64,64) f32
  const float* target = (const float*)d_in[1];       // (32,256,256)   f32
  float* out          = (float*)d_out;               // scalar f32
  __hip_bfloat16* F   = (__hip_bfloat16*)d_ws;       // 32*256*4096 bf16 = 64 MB

  hipMemsetAsync(out, 0, sizeof(float), stream);     // capture-safe memset node

  topk_mask_kernel<<<NROWS, 256, 0, stream>>>(x, F);

  // 256 blocks x 8 waves: one 128x64 G tile per block, one 32x32 per wave
  gram_loss_kernel<<<256, 256, 0, stream>>>(F, target, out);
}
